// WLSKernelLayer_49065706389958
// MI455X (gfx1250) — matrix operand order repaired and checked by reading, not compiled, not run on hardware
//
#include <hip/hip_runtime.h>

// ---------------------------------------------------------------------------
// WLS kernel layer for MI455X (gfx1250, wave32, WMMA).
//
// Math: out = ((segment_sum_{dst}(h[src]) + h) @ proj) * (1/sqrt(16))
//       with h = [1, sqrt(2)*z, vec(z z^T)], z = 0.1*x  (dim 273).
// Projection is linear and the +/-1e6 clips never bind at these magnitudes
// (|h| <= 1, fan-in ~8 edges), so we project FIRST:
//       y    = h @ proj                       (WMMA GEMM, [N,273]x[273,64])
//       out  = (y + segsum_{dst}(y[src])) / 4 (f32 atomics + scale)
// One wave computes a full 16x64 output stripe: the phi(z) A-fragments
// (72 VGPRs) are built once and reused across all 4 col-tiles.
// ---------------------------------------------------------------------------

typedef __attribute__((ext_vector_type(16))) _Float16 v16h;
typedef __attribute__((ext_vector_type(8)))  float    v8f;
typedef __attribute__((ext_vector_type(4)))  float    v4f;

#define IN_DIM   16
#define OUT_DIM  64
#define EXP_DIM  273            // 1 + 16 + 256
#define KSTEPS   9              // K padded to 288 = 9 * 32
#define NCT      4              // 4 col tiles of 16 = 64 output cols
#define SQRT2C   1.41421356237f
#define FSCALE   0.1f
#define INV_SQRT_IN 0.25f

// phi(z)[k] with k a compile-time constant after full unrolling -> all z
// indices constant -> z stays in VGPRs (no scratch spills).
__device__ __forceinline__ float hval(const float z[IN_DIM], int k) {
    if (k == 0)        return 1.0f;
    if (k <= IN_DIM)   return SQRT2C * z[k - 1];
    if (k <  EXP_DIM) { int t = k - 17; return z[t >> 4] * z[t & 15]; }
    return 0.0f;       // K padding 273..287
}

// ---------------------------------------------------------------------------
// Kernel 1: y = h @ proj via v_wmma_f32_16x16x32_f16.
// One wave == one 16-row x 64-col output stripe (4 x 16x16 WMMA tiles,
// 9 K-steps each -> 36 WMMAs per wave, A reused across col tiles).
// ---------------------------------------------------------------------------
__global__ __launch_bounds__(256) void wls_gemm_kernel(
    const float* __restrict__ features,   // [n_nodes, 16]
    const float* __restrict__ proj,       // [273, 64]
    float* __restrict__ y_ws,             // [n_nodes, 64] scratch copy of y
    float* __restrict__ acc,              // [n_nodes, 64] accumulator (d_out)
    int n_row_tiles)
{
    const int wave_in_blk = threadIdx.x >> 5;
    const int rt          = blockIdx.x * 8 + wave_in_blk;   // row tile
    if (rt >= n_row_tiles) return;        // wave-uniform exit, no barriers used

    const int lane = threadIdx.x & 31;
    const int rloc = lane & 15;           // M = lane % 16
    const int half = lane >> 4;           // lane group within the wave

    // Load this lane's z row (16 f32, constant indices -> registers).
    float z[IN_DIM];
    const float* frow = features + (size_t)(rt * 16 + rloc) * IN_DIM;
    {
        v4f z0 = *(const v4f*)(frow + 0);
        v4f z1 = *(const v4f*)(frow + 4);
        v4f z2 = *(const v4f*)(frow + 8);
        v4f z3 = *(const v4f*)(frow + 12);
#pragma unroll
        for (int i = 0; i < 4; ++i) {
            z[i]      = FSCALE * z0[i];
            z[i + 4]  = FSCALE * z1[i];
            z[i + 8]  = FSCALE * z2[i];
            z[i + 12] = FSCALE * z3[i];
        }
    }

    // Build all 9 A-fragments up front. The K index depends on `half`, so
    // branch once per lane group and keep k compile-time constant inside each
    // branch (constant z[] indices -> registers). Divergence ends here; all
    // WMMAs below run with EXEC all-ones.
    v16h A[KSTEPS];
    if (half == 0) {
#pragma unroll
        for (int s = 0; s < KSTEPS; ++s)
#pragma unroll
            for (int j = 0; j < 16; ++j) {
                int k = 32 * s + (j & 7) + ((j & 8) ? 16 : 0);      // +0
                A[s][j] = (_Float16)hval(z, k);
            }
    } else {
#pragma unroll
        for (int s = 0; s < KSTEPS; ++s)
#pragma unroll
            for (int j = 0; j < 16; ++j) {
                int k = 32 * s + (j & 7) + ((j & 8) ? 16 : 0) + 8;  // +8
                A[s][j] = (_Float16)hval(z, k);
            }
    }

    // 4 accumulators, one per 16-col tile; A fragments reused 4x.
    v8f C[NCT];
#pragma unroll
    for (int ct = 0; ct < NCT; ++ct) C[ct] = (v8f){};

#pragma unroll
    for (int s = 0; s < KSTEPS; ++s) {
#pragma unroll
        for (int ct = 0; ct < NCT; ++ct) {
            // B fragment: lane holds column ct*16 + lane%16,
            // K = 32s + 8*half + (j&7) + (j&8 ? 16 : 0).
            const int colg = ct * 16 + rloc;
            v16h b;
#pragma unroll
            for (int j = 0; j < 16; ++j) {
                int k = 32 * s + 8 * half + (j & 7) + ((j & 8) ? 16 : 0);
                float pv = (k < EXP_DIM) ? proj[(size_t)k * OUT_DIM + colg]
                                         : 0.0f;
                b[j] = (_Float16)pv;
            }
            C[ct] = __builtin_amdgcn_wmma_f32_16x16x32_f16(
                        /*neg_a=*/false, A[s], /*neg_b=*/false, b,
                        /*c_mod=*/(short)0, C[ct],
                        /*reuse_a=*/false, /*reuse_b=*/false);
        }
    }

    // C/D layout: VGPR r -> row half*8 + r, col = lane % 16 within tile.
#pragma unroll
    for (int ct = 0; ct < NCT; ++ct) {
        const int colg = ct * 16 + rloc;
#pragma unroll
        for (int r = 0; r < 8; ++r) {
            int rowg = rt * 16 + half * 8 + r;
            size_t off = (size_t)rowg * OUT_DIM + colg;
            float v = C[ct][r];
            y_ws[off] = v;    // source for edge gather
            acc[off]  = v;    // residual term; scatter adds on top of this
        }
    }
}

// ---------------------------------------------------------------------------
// Kernel 2: edge scatter — acc[dst] += y[src], 64 lanes per edge.
// ---------------------------------------------------------------------------
__global__ __launch_bounds__(256) void wls_scatter_kernel(
    const int* __restrict__ src, const int* __restrict__ dst,
    const float* __restrict__ y_ws, float* __restrict__ acc, int n_edges)
{
    const int e = blockIdx.x * 4 + (threadIdx.x >> 6);
    const int j = threadIdx.x & 63;
    if (e >= n_edges) return;
    const int s = src[e];
    const int d = dst[e];
    atomicAdd(&acc[(size_t)d * OUT_DIM + j], y_ws[(size_t)s * OUT_DIM + j]);
}

// ---------------------------------------------------------------------------
// Kernel 3: out *= 1/sqrt(16), vectorized 4-wide.
// ---------------------------------------------------------------------------
__global__ __launch_bounds__(256) void wls_finalize_kernel(
    float* __restrict__ out, int n4)
{
    const int i = blockIdx.x * blockDim.x + threadIdx.x;
    if (i >= n4) return;
    v4f v = ((v4f*)out)[i];
    v *= INV_SQRT_IN;
    ((v4f*)out)[i] = v;
}

// ---------------------------------------------------------------------------
extern "C" void kernel_launch(void* const* d_in, const int* in_sizes, int n_in,
                              void* d_out, int out_size, void* d_ws, size_t ws_size,
                              hipStream_t stream)
{
    const float* features = (const float*)d_in[0];
    const int*   src      = (const int*)  d_in[1];
    const int*   dst      = (const int*)  d_in[2];
    const float* proj     = (const float*)d_in[3];

    const int n_nodes = in_sizes[0] / IN_DIM;   // 50000 (multiple of 16)
    const int n_edges = in_sizes[1];            // 400000

    float* y_ws = (float*)d_ws;                 // n_nodes*64 f32 = 12.8 MB
    float* acc  = (float*)d_out;

    // 1) GEMM: y = phi(z) @ proj  -> y_ws and acc (residual term).
    const int n_row_tiles = n_nodes / 16;
    const int gemm_blocks = (n_row_tiles + 7) / 8;  // 8 waves per block
    wls_gemm_kernel<<<gemm_blocks, 256, 0, stream>>>(
        features, proj, y_ws, acc, n_row_tiles);

    // 2) Scatter-add projected rows over edges.
    const int scat_blocks = (n_edges + 3) / 4;  // 4 edges per block (64 thr/edge)
    wls_scatter_kernel<<<scat_blocks, 256, 0, stream>>>(
        src, dst, y_ws, acc, n_edges);

    // 3) Scale by 1/sqrt(in_dim).
    const int n4 = (n_nodes * OUT_DIM) / 4;
    wls_finalize_kernel<<<(n4 + 255) / 256, 256, 0, stream>>>(acc, n4);
}
